// ConvNAT_506806141141
// MI455X (gfx1250) — matrix operand order repaired and checked by reading, not compiled, not run on hardware
//
#include <hip/hip_runtime.h>
#include <hip/hip_bf16.h>

typedef __attribute__((ext_vector_type(16))) _Float16 v16h;
typedef __attribute__((ext_vector_type(8)))  _Float16 v8h;
typedef __attribute__((ext_vector_type(8)))  float    v8f;

#define BB   4
#define HH   56
#define WW   56
#define CC   128
#define HIM  112
#define WIM  112
#define WIN  7
#define PW   32          // padded patch width (>= 22)
#define PR   7           // patch rows
#define KP   (PR*PW)     // 224
#define NPIX (BB*HH*WW)  // 12544

// ---- A-matrix (16x32 f16) lane loader, contiguous k layout ----
// lane<16 : halves[0..15]=K(k0+0..15)
// lane>=16: halves[0..15]=K(k0+16..31)
__device__ __forceinline__ v16h load_a16(const _Float16* row, int k0, int lane) {
  int off = (lane & 16) ? 16 : 0;
  v8h lo = *(const v8h*)(row + k0 + off);
  v8h hi = *(const v8h*)(row + k0 + off + 8);
  return __builtin_shufflevector(lo, hi, 0,1,2,3,4,5,6,7,8,9,10,11,12,13,14,15);
}

// ---------------- weight fp32 -> f16 ----------------
__global__ void cvt_w_kernel(const float* __restrict__ qw, const float* __restrict__ kw,
                             const float* __restrict__ vw, _Float16* __restrict__ w16) {
  int i = blockIdx.x * blockDim.x + threadIdx.x;
  if (i >= 3 * CC * CC) return;
  int mat = i / (CC * CC), off = i % (CC * CC);
  const float* s = (mat == 0) ? qw : (mat == 1) ? kw : vw;
  w16[i] = (_Float16)s[off];
}

// ---------------- conv3x3 s2 + bias + relu -> f16 NHWC ----------------
__global__ void conv_relu_kernel(const float* __restrict__ x, const float* __restrict__ w,
                                 const float* __restrict__ bias, _Float16* __restrict__ f16) {
  int pix = blockIdx.x;
  int c   = threadIdx.x;
  int b   = pix / (HH * WW);
  int r   = pix % (HH * WW);
  int h = r / WW, wo = r % WW;
  float acc = bias[c];
  const float* wc = w + c * 27;
  for (int ci = 0; ci < 3; ++ci)
    for (int kh = 0; kh < 3; ++kh) {
      int ih = 2 * h - 1 + kh;
      if (ih < 0 || ih >= HIM) continue;
      for (int kw2 = 0; kw2 < 3; ++kw2) {
        int iw = 2 * wo - 1 + kw2;
        if (iw < 0 || iw >= WIM) continue;
        acc += x[((b * 3 + ci) * HIM + ih) * WIM + iw] * wc[(ci * 3 + kh) * 3 + kw2];
      }
    }
  f16[(size_t)pix * CC + c] = (_Float16)fmaxf(acc, 0.f);
}

// ---------------- QKV projections via WMMA ----------------
// grid (784 Mtiles, 3 projections), block 256 = 8 waves, wave = N-tile
__global__ void qkv_wmma_kernel(const _Float16* __restrict__ f16, const _Float16* __restrict__ w16,
                                const float* __restrict__ qb, const float* __restrict__ kb,
                                const float* __restrict__ vb,
                                _Float16* __restrict__ q16, _Float16* __restrict__ k16,
                                _Float16* __restrict__ v16) {
  int mt   = blockIdx.x;
  int p    = blockIdx.y;
  int lane = threadIdx.x & 31;
  int nt   = threadIdx.x >> 5;   // 0..7
  int m    = lane & 15;
  int n    = nt * 16 + m;
  const _Float16* A    = f16 + (size_t)(mt * 16 + m) * CC;
  const _Float16* Wrow = w16 + (size_t)p * CC * CC + (size_t)n * CC;
  int boff = (lane & 16) ? 16 : 0;
  v8f acc = {};
#pragma unroll
  for (int k0 = 0; k0 < CC; k0 += 32) {
    v16h a  = load_a16(A, k0, lane);
    v16h bm = *(const v16h*)(Wrow + k0 + boff);
    acc = __builtin_amdgcn_wmma_f32_16x16x32_f16(false, a, false, bm, (short)0, acc, false, false);
  }
  const float* bias = (p == 0) ? qb : (p == 1) ? kb : vb;
  _Float16* outp    = (p == 0) ? q16 : (p == 1) ? k16 : v16;
  float scale = (p == 0) ? 0.08838834764831845f : 1.0f;  // C^-0.5 folded into q
  float bv = bias[n];
  int mbase = mt * 16 + ((lane & 16) ? 8 : 0);
#pragma unroll
  for (int r = 0; r < 8; ++r)
    outp[(size_t)(mbase + r) * CC + n] = (_Float16)((acc[r] + bv) * scale);
}

// ---------------- neighborhood attention via WMMA ----------------
// grid (4 col-tiles, 56 rows, 4 batch), block = 32 (one wave = 16 pixels)
__global__ void nat_attn_kernel(const _Float16* __restrict__ q16, const _Float16* __restrict__ k16,
                                const _Float16* __restrict__ v16, float* __restrict__ out) {
  __shared__ alignas(32) float    logitsL[16][KP];
  __shared__ alignas(32) _Float16 attnL[16][KP];
  __shared__ alignas(32) _Float16 Vt[16][240];   // 16 channels x 224 neighbors (+pad)

  int ct = blockIdx.x, h = blockIdx.y, b = blockIdx.z;
  int lane = threadIdx.x & 31;
  int m    = lane & 15;
  int hi16 = (lane & 16) ? 1 : 0;
  int boff = hi16 ? 16 : 0;

  int c0 = (ct < 3) ? ct * 16 : 40;                        // tile starts 0,16,32,40 (overlap)
  int rs = h - 3;  rs = rs < 0 ? 0 : (rs > HH - WIN ? HH - WIN : rs);
  int cb = c0 - 3; cb = cb < 0 ? 0 : (cb > WW - WIN ? WW - WIN : cb);

  // q A-fragments: load once, keep in VGPRs for all 14 logits WMMA tiles
  const _Float16* A = q16 + (size_t)((b * HH + h) * WW + c0 + m) * CC;
  v16h aq[4];
#pragma unroll
  for (int t = 0; t < 4; ++t) aq[t] = load_a16(A, t * 32, lane);

  // ---- logits: 16 pixels x (7 rows x 32 cols) patch ----
  for (int nr = 0; nr < PR; ++nr) {
    int prow = (b * HH + rs + nr) * WW;
    for (int jt = 0; jt < 2; ++jt) {
      int j   = jt * 16 + m;
      int col = cb + j;
      bool ok = (col < WW);
      const _Float16* Brow = k16 + (size_t)(prow + col) * CC;
      v8f acc = {};
#pragma unroll
      for (int t = 0; t < 4; ++t) {
        v16h bm = {};
        if (ok) bm = *(const v16h*)(Brow + t * 32 + boff);
        acc = __builtin_amdgcn_wmma_f32_16x16x32_f16(false, aq[t], false, bm, (short)0, acc, false, false);
      }
      int kk = nr * PW + j;
      int mb = hi16 ? 8 : 0;
#pragma unroll
      for (int r = 0; r < 8; ++r) logitsL[mb + r][kk] = acc[r];
    }
  }
  __syncthreads();

  // ---- per-pixel softmax over the 49 valid neighbors ----
  if (lane < 16) {
    int c  = c0 + lane;
    int cs = c - 3; cs = cs < 0 ? 0 : (cs > WW - WIN ? WW - WIN : cs);
    int jlo = cs - cb;                     // valid cols = [jlo, jlo+6]
    float mx = -3.0e38f;
    for (int pr = 0; pr < PR; ++pr)
      for (int j = jlo; j < jlo + WIN; ++j)
        mx = fmaxf(mx, logitsL[lane][pr * PW + j]);
    float s = 0.f;
    for (int pr = 0; pr < PR; ++pr)
      for (int j = jlo; j < jlo + WIN; ++j)
        s += __expf(logitsL[lane][pr * PW + j] - mx);
    float inv = 1.0f / s;
    for (int kk = 0; kk < KP; ++kk) {
      int j = kk & (PW - 1);
      float a = (j >= jlo && j < jlo + WIN) ? __expf(logitsL[lane][kk] - mx) * inv : 0.f;
      attnL[lane][kk] = (_Float16)a;
    }
  }
  __syncthreads();

  // attn A-fragments: load once from LDS, reuse across all 8 channel tiles
  v16h aat[7];
#pragma unroll
  for (int t = 0; t < 7; ++t) aat[t] = load_a16(&attnL[m][0], t * 32, lane);

  // ---- out = attn (16x224) x V (224x128), 16 channels per pass ----
  for (int nt2 = 0; nt2 < 8; ++nt2) {
    __syncthreads();
    // transpose this channel slice of the V patch into LDS
    for (int t = 0; t < 7; ++t) {
      int kk = t * 32 + lane;
      int pr = kk >> 5, j = kk & 31;
      int col = cb + j;
      if (col < WW) {
        const _Float16* vr = v16 + (size_t)((b * HH + rs + pr) * WW + col) * CC + nt2 * 16;
        v16h vv = *(const v16h*)vr;
#pragma unroll
        for (int cch = 0; cch < 16; ++cch) Vt[cch][kk] = vv[cch];
      } else {
#pragma unroll
        for (int cch = 0; cch < 16; ++cch) Vt[cch][kk] = (_Float16)0.f;
      }
    }
    __syncthreads();
    v8f acc = {};
#pragma unroll
    for (int t = 0; t < 7; ++t) {
      v16h bm = *(const v16h*)(&Vt[m][0] + t * 32 + boff);
      acc = __builtin_amdgcn_wmma_f32_16x16x32_f16(false, aat[t], false, bm, (short)0, acc, false, false);
    }
    // tile 3 overlaps tile 2 on cols 40..47: only its hi16 half (48..55) stores,
    // so every output element has exactly one writer (deterministic).
    if (ct < 3 || hi16) {
      int n = nt2 * 16 + m;
      int cbase = c0 + (hi16 ? 8 : 0);
#pragma unroll
      for (int r = 0; r < 8; ++r)
        out[(size_t)((b * CC + n) * HH + h) * WW + cbase + r] = acc[r];
    }
  }
}

extern "C" void kernel_launch(void* const* d_in, const int* in_sizes, int n_in,
                              void* d_out, int out_size, void* d_ws, size_t ws_size,
                              hipStream_t stream) {
  (void)in_sizes; (void)n_in; (void)out_size; (void)ws_size;
  const float* x      = (const float*)d_in[0];
  const float* conv_w = (const float*)d_in[1];
  const float* conv_b = (const float*)d_in[2];
  const float* q_w    = (const float*)d_in[3];
  const float* q_b    = (const float*)d_in[4];
  const float* k_w    = (const float*)d_in[5];
  const float* k_b    = (const float*)d_in[6];
  const float* v_w    = (const float*)d_in[7];
  const float* v_b    = (const float*)d_in[8];
  float* out = (float*)d_out;

  char* ws = (char*)d_ws;
  const size_t SZ = (size_t)NPIX * CC * sizeof(_Float16);  // 3,211,264 B (32B-aligned)
  _Float16* f16 = (_Float16*)(ws);
  _Float16* q16 = (_Float16*)(ws + SZ);
  _Float16* k16 = (_Float16*)(ws + 2 * SZ);
  _Float16* v16 = (_Float16*)(ws + 3 * SZ);
  _Float16* w16 = (_Float16*)(ws + 4 * SZ);                // 3*128*128 halves

  cvt_w_kernel<<<(3 * CC * CC + 255) / 256, 256, 0, stream>>>(q_w, k_w, v_w, w16);
  conv_relu_kernel<<<NPIX, CC, 0, stream>>>(x, conv_w, conv_b, f16);
  qkv_wmma_kernel<<<dim3(NPIX / 16, 3), 256, 0, stream>>>(f16, w16, q_b, k_b, v_b, q16, k16, v16);
  nat_attn_kernel<<<dim3(4, HH, BB), 32, 0, stream>>>(q16, k16, v16, out);
}
